// DissipativeRINN_22582938042515
// MI455X (gfx1250) — matrix operand order repaired and checked by reading, not compiled, not run on hardware
//
#include <hip/hip_runtime.h>
#include <cmath>

typedef __attribute__((ext_vector_type(16))) _Float16 v16h;
typedef __attribute__((ext_vector_type(8)))  _Float16 v8h;
typedef __attribute__((ext_vector_type(8)))  float    v8f;
typedef __attribute__((ext_vector_type(4)))  float    v4f;

namespace {
constexpr int kB = 2048, kT = 32, kIN = 16, kST = 16, kNL = 128, kOUT = 8, kCH = 17;
constexpr float kDT = 0.01f;
constexpr int kFullIters = 30, kReuseIters = 5;
constexpr int kWaves = 2, kRows = 16;

// All B-operand weight matrices stored column-major f16 in LDS so a wave32
// B-fragment (32x16 / 128x16 K-chunk) is two contiguous 16B LDS reads per lane.
struct __align__(32) Smem {
  _Float16 dvw  [kNL * kNL];   // [col*128 + k]   Dvw_T col-major
  _Float16 cvdvy[kNL * 32];    // [col*32 + k]    rows 0..15 = Cv_T, 16..31 = Dvy_T
  _Float16 bw   [16 * kNL];    // [col*128 + k]   Bw_T col-major
  _Float16 duw  [16 * kNL];    // [col*128 + k]   Duw_T col-major, cols 8..15 zero-pad
  _Float16 ayby [16 * 32];     // [col*32 + k]    rows 0..15 = A_T, 16..31 = By_T
  _Float16 cuduy[16 * 32];     // [col*32 + k]    rows = [Cu_T; Duy_T], cols 8..15 zero
  _Float16 xy   [kWaves][16 * 32];    // per wave, row-major: cols 0..15 = x, 16..31 = y
  _Float16 wbuf [kWaves][16 * kNL];   // per wave, row-major 16x128 f16 w iterate
};
}  // namespace

// gfx1250 has a hardware TRANS32 tanh (V_TANH_F32); use it when the toolchain
// exposes the builtin. Falls back to libm tanhf otherwise (compile-safe).
__device__ __forceinline__ float fast_tanh(float x) {
#if __has_builtin(__builtin_amdgcn_tanhf)
  return __builtin_amdgcn_tanhf(x);
#elif __has_builtin(__builtin_amdgcn_tanh_f32)
  return __builtin_amdgcn_tanh_f32(x);
#else
  return tanhf(x);
#endif
}

// Wave-local LDS publish: xy/wbuf tiles are private to one wave, so cross-lane
// store->load visibility only needs wave-local ordering. DS ops from one wave
// complete in order (DScnt, in-order per ISA); the wait + compiler memory
// fence replaces a full s_barrier pair and decouples the two waves of a block.
__device__ __forceinline__ void lds_fence_wave() {
  asm volatile("s_wait_dscnt 0x0" ::: "memory");
}

__device__ __forceinline__ v8f vzero() {
  v8f z;
#pragma unroll
  for (int i = 0; i < 8; ++i) z[i] = 0.0f;
  return z;
}

// A fragment (16xK row-major f16 tile in LDS), K-chunk k0..k0+31.
// lane<16: halves 0..7 = K k0..k0+7, halves 8..15 = K k0+16..k0+23 (row = lane)
// lane>=16: K k0+8..k0+15 and K k0+24..k0+31 (row = lane-16)
__device__ __forceinline__ v16h load_a_frag(const _Float16* tile, int stride, int k0, int lane) {
  const int row = lane & 15;
  const int h   = lane >> 4;
  const _Float16* p = tile + row * stride + k0 + h * 8;
  v8h lo = *(const v8h*)(p);
  v8h hi = *(const v8h*)(p + 16);
  return __builtin_shufflevector(lo, hi, 0,1,2,3,4,5,6,7,8,9,10,11,12,13,14,15);
}

// B fragment (Kx16 col-major f16 in LDS): lane<16 holds col, K k0..k0+7 and k0+16..k0+23;
// lane>=16 holds col, K k0+8..k0+15 and k0+24..k0+31 -> two 16B reads per lane.
__device__ __forceinline__ v16h load_b_frag(const _Float16* tile_cm, int kTot, int k0, int col0, int lane) {
  const int colj = col0 + (lane & 15);
  const int ks   = lane >> 4;
  const _Float16* p = tile_cm + colj * kTot + k0 + ks * 8;
  v8h lo = *(const v8h*)(p);
  v8h hi = *(const v8h*)(p + 16);
  return __builtin_shufflevector(lo, hi, 0,1,2,3,4,5,6,7,8,9,10,11,12,13,14,15);
}

__device__ __forceinline__ v8f wmma_f16(v16h a, v16h b, v8f c) {
  // 8 args: (neg_a, A, neg_b, B, c_mod, C, reuse_a, reuse_b)
  return __builtin_amdgcn_wmma_f32_16x16x32_f16(false, a, false, b, (short)0, c, false, false);
}

// C-fragment (16x16 f32: lane holds col N=lane%16, vgpr v holds row v+8*(lane/16))
// -> row-major f16 x-columns of the [x|y] tile.
__device__ __forceinline__ void store_xtile(const v8f& x, _Float16* xy, int lane) {
  const int colj = lane & 15, h = lane >> 4;
#pragma unroll
  for (int v = 0; v < 8; ++v) xy[(v + 8 * h) * 32 + colj] = (_Float16)x[v];
}

// One xdot: base = [x|y]@[Cv;Dvy]; w <- tanh(base + w@Dvw) x iters;
// k = [x|y]@[A;By] + w@Bw. wl holds warm-start on entry, solution on exit.
__device__ __forceinline__ void run_xdot(const v8f& xeff, int iters,
                                         _Float16* xy, _Float16* wl, const Smem& sm,
                                         const v16h (*dvwf)[8], int lane,
                                         v8f& k_out, v16h& axy_out) {
  store_xtile(xeff, xy, lane);
  lds_fence_wave();                    // publish x (and any prior y / w0 writes)
  const v16h axy = load_a_frag(xy, 32, 0, lane);

  v8f base[8];
#pragma unroll
  for (int n = 0; n < 8; ++n)
    base[n] = wmma_f16(axy, load_b_frag(sm.cvdvy, 32, 0, n * 16, lane), vzero());

  const int colj = lane & 15, h = lane >> 4;
  for (int it = 0; it < iters; ++it) {
    v8f c[8];
#pragma unroll
    for (int n = 0; n < 8; ++n) c[n] = base[n];
#pragma unroll
    for (int k = 0; k < 4; ++k) {
      const v16h a = load_a_frag(wl, kNL, k * 32, lane);
#pragma unroll
      for (int n = 0; n < 8; ++n) c[n] = wmma_f16(a, dvwf[k][n], c[n]);
    }
    lds_fence_wave();                  // all A-frag reads done before overwrite
#pragma unroll
    for (int n = 0; n < 8; ++n) {
#pragma unroll
      for (int v = 0; v < 8; ++v)
        wl[(v + 8 * h) * kNL + n * 16 + colj] = (_Float16)fast_tanh(c[n][v]);
    }
    lds_fence_wave();                  // publish w iterate for next round
  }

  v8f kc = wmma_f16(axy, load_b_frag(sm.ayby, 32, 0, 0, lane), vzero());
#pragma unroll
  for (int k = 0; k < 4; ++k)
    kc = wmma_f16(load_a_frag(wl, kNL, k * 32, lane),
                  load_b_frag(sm.bw, kNL, k * 32, 0, lane), kc);
  k_out = kc;
  axy_out = axy;
}

__global__ __launch_bounds__(kWaves * 32) void rinn_rollout(
    const float* __restrict__ obs, const float* __restrict__ x0,
    const float* __restrict__ A_T, const float* __restrict__ Bw_T,
    const float* __restrict__ By_T, const float* __restrict__ Cv_T,
    const float* __restrict__ Dvw_T, const float* __restrict__ Dvy_T,
    const float* __restrict__ Cu_T, const float* __restrict__ Duw_T,
    const float* __restrict__ Duy_T, const float* __restrict__ log_stds,
    float* __restrict__ out) {
  __shared__ Smem sm;
  const int tid = threadIdx.x;
  const int lane = tid & 31;
  const int wv = tid >> 5;

  // ---- stage weights (f32 global -> f16 LDS, B-operand col-major) ----
  for (int i = tid; i < kNL * kNL; i += blockDim.x) {
    int colj = i >> 7, k = i & 127;
    sm.dvw[i] = (_Float16)Dvw_T[k * kNL + colj];
  }
  for (int i = tid; i < kNL * 32; i += blockDim.x) {
    int colj = i >> 5, k = i & 31;
    float v = (k < 16) ? Cv_T[k * kNL + colj] : Dvy_T[(k - 16) * kNL + colj];
    sm.cvdvy[i] = (_Float16)v;
  }
  for (int i = tid; i < 16 * kNL; i += blockDim.x) {
    int colj = i >> 7, k = i & 127;
    sm.bw[i]  = (_Float16)Bw_T[k * 16 + colj];
    sm.duw[i] = (_Float16)((colj < kOUT) ? Duw_T[k * kOUT + colj] : 0.0f);
  }
  for (int i = tid; i < 16 * 32; i += blockDim.x) {
    int colj = i >> 5, k = i & 31;
    float va = (k < 16) ? A_T[k * 16 + colj] : By_T[(k - 16) * 16 + colj];
    sm.ayby[i] = (_Float16)va;
    float vu = 0.0f;
    if (colj < kOUT) vu = (k < 16) ? Cu_T[k * kOUT + colj] : Duy_T[(k - 16) * kOUT + colj];
    sm.cuduy[i] = (_Float16)vu;
  }
  __syncthreads();   // the only cross-wave dependency: staged weights

  // Dvw B-fragments -> registers; reused 45x per step (hot loop never touches
  // LDS for B). 32 frags * 8 VGPRs = 256 VGPRs, within CDNA5's 1024/wave.
  v16h dvwf[4][8];
#pragma unroll
  for (int k = 0; k < 4; ++k)
#pragma unroll
    for (int n = 0; n < 8; ++n)
      dvwf[k][n] = load_b_frag(sm.dvw, kNL, k * 32, n * 16, lane);

  const int b0 = (blockIdx.x * kWaves + wv) * kRows;
  _Float16* xy = sm.xy[wv];
  _Float16* wl = sm.wbuf[wv];
  const int colj = lane & 15, h = lane >> 4;

  // x state in C-fragment layout, persists across all T steps
  v8f x;
#pragma unroll
  for (int v = 0; v < 8; ++v) x[v] = x0[(size_t)(b0 + v + 8 * h) * kST + colj];

  const float lsd = (colj >= kOUT) ? log_stds[colj - kOUT] : 0.0f;

  for (int t = 0; t < kT; ++t) {
    // y -> xy cols 16..31 (f16); each lane: 8 contiguous f32 -> one 16B LDS store
    {
      const int row = lane >> 1, half = lane & 1;
      const float* src = obs + ((size_t)(b0 + row) * kT + t) * kIN + half * 8;
      v4f y0 = *(const v4f*)(src);
      v4f y1 = *(const v4f*)(src + 4);
      v8h yh;
#pragma unroll
      for (int i = 0; i < 4; ++i) { yh[i] = (_Float16)y0[i]; yh[4 + i] = (_Float16)y1[i]; }
      *(v8h*)(xy + row * 32 + 16 + half * 8) = yh;
    }
    // w0 = 0 (cold start for the 30-iter solve)
    {
      v8h z8;
#pragma unroll
      for (int i = 0; i < 8; ++i) z8[i] = (_Float16)0.0f;
#pragma unroll
      for (int j = 0; j < 8; ++j) *(v8h*)(wl + lane * 64 + j * 8) = z8;
    }
    // (store_xtile + wave fence inside run_xdot publishes y & w0 too)

    v8f k1, k2, k3, k4; v16h axy1, axyt;
    run_xdot(x, kFullIters, xy, wl, sm, dvwf, lane, k1, axy1);

    // u = x@Cu_T + w1@Duw_T + y@Duy_T  (uses original x via axy1, w1 in wl)
    v8f uc = wmma_f16(axy1, load_b_frag(sm.cuduy, 32, 0, 0, lane), vzero());
#pragma unroll
    for (int k = 0; k < 4; ++k)
      uc = wmma_f16(load_a_frag(wl, kNL, k * 32, lane),
                    load_b_frag(sm.duw, kNL, k * 32, 0, lane), uc);
#pragma unroll
    for (int v = 0; v < 8; ++v) {
      size_t o = ((size_t)(b0 + v + 8 * h) * kT + t) * kCH + colj;
      out[o] = (colj < kOUT) ? uc[v] : lsd;   // channels 0..7 means, 8..15 log_std
    }

    run_xdot(x + k1 * (kDT * 0.5f), kReuseIters, xy, wl, sm, dvwf, lane, k2, axyt);
    run_xdot(x + k2 * (kDT * 0.5f), kReuseIters, xy, wl, sm, dvwf, lane, k3, axyt);
    run_xdot(x + k3 * kDT,          kReuseIters, xy, wl, sm, dvwf, lane, k4, axyt);
    x = x + (k1 + k2 * 2.0f + k3 * 2.0f + k4) * (kDT / 6.0f);
  }
}

// Value baseline MLP: one thread per (b,t) row; weights broadcast from LDS.
__global__ __launch_bounds__(256) void rinn_value(
    const float* __restrict__ obs, const float* __restrict__ W1,
    const float* __restrict__ b1, const float* __restrict__ W2,
    const float* __restrict__ b2, const float* __restrict__ W3,
    const float* __restrict__ b3, float* __restrict__ out) {
  __shared__ float sW1[16 * 64];
  __shared__ float sW2[64 * 64];
  __shared__ float sb1[64], sb2[64], sW3[64];
  __shared__ float sb3;
  const int tid = threadIdx.x;
  for (int i = tid; i < 16 * 64; i += 256) sW1[i] = W1[i];
  for (int i = tid; i < 64 * 64; i += 256) sW2[i] = W2[i];
  if (tid < 64) { sb1[tid] = b1[tid]; sb2[tid] = b2[tid]; sW3[tid] = W3[tid]; }
  if (tid == 0) sb3 = b3[0];
  __syncthreads();

  const size_t r = (size_t)blockIdx.x * 256 + tid;
  if (r >= (size_t)kB * kT) return;
  float o[16];
  const float* op = obs + r * 16;
#pragma unroll
  for (int i = 0; i < 16; ++i) o[i] = op[i];
  float hbuf[64];
#pragma unroll 4
  for (int j = 0; j < 64; ++j) {
    float a = sb1[j];
#pragma unroll
    for (int i = 0; i < 16; ++i) a = fmaf(o[i], sW1[i * 64 + j], a);
    hbuf[j] = fast_tanh(a);
  }
  float val = sb3;
#pragma unroll 2
  for (int j = 0; j < 64; ++j) {
    float a = sb2[j];
    for (int i = 0; i < 64; ++i) a = fmaf(hbuf[i], sW2[i * 64 + j], a);
    val = fmaf(fast_tanh(a), sW3[j], val);
  }
  out[r * kCH + 16] = val;
}

extern "C" void kernel_launch(void* const* d_in, const int* in_sizes, int n_in,
                              void* d_out, int out_size, void* d_ws, size_t ws_size,
                              hipStream_t stream) {
  const float* obs      = (const float*)d_in[0];
  const float* x0       = (const float*)d_in[1];
  const float* A_T      = (const float*)d_in[2];
  const float* Bw_T     = (const float*)d_in[3];
  const float* By_T     = (const float*)d_in[4];
  const float* Cv_T     = (const float*)d_in[5];
  const float* Dvw_T    = (const float*)d_in[6];
  const float* Dvy_T    = (const float*)d_in[7];
  const float* Cu_T     = (const float*)d_in[8];
  const float* Duw_T    = (const float*)d_in[9];
  const float* Duy_T    = (const float*)d_in[10];
  const float* log_stds = (const float*)d_in[11];
  const float* W1       = (const float*)d_in[12];
  const float* b1       = (const float*)d_in[13];
  const float* W2       = (const float*)d_in[14];
  const float* b2       = (const float*)d_in[15];
  const float* W3       = (const float*)d_in[16];
  const float* b3       = (const float*)d_in[17];
  float* out = (float*)d_out;

  (void)in_sizes; (void)n_in; (void)out_size; (void)d_ws; (void)ws_size;

  rinn_rollout<<<dim3(kB / (kWaves * kRows)), dim3(kWaves * 32), 0, stream>>>(
      obs, x0, A_T, Bw_T, By_T, Cv_T, Dvw_T, Dvy_T, Cu_T, Duw_T, Duy_T, log_stds, out);

  rinn_value<<<dim3((kB * kT) / 256), dim3(256), 0, stream>>>(
      obs, W1, b1, W2, b2, W3, b3, out);
}
